// global_MDTA_14989435863612
// MI455X (gfx1250) — matrix operand order repaired and checked by reading, not compiled, not run on hardware
//
#include <hip/hip_runtime.h>
#include <hip/hip_bf16.h>

// ---------------------------------------------------------------------------
// MDTA-style block: qkv 1x1 conv -> 3x3 depthwise -> spatial attention
// (4 heads, 4096 tokens, head_dim 32) -> 1x1 projection.
// Attention dominates (~8.6 GFLOP); flash attention with
// v_wmma_f32_16x16x32_bf16, f32 online softmax, DPP row reductions,
// double-buffered K/V tiles (one barrier per key block).
// ---------------------------------------------------------------------------

#define N_TOK 4096
#define HD    32
#define HEADS 4
#define KB    64      // key-block per iteration
#define QW    16      // queries per wave
#define WAVES 8       // waves per block (256 threads, wave32)

typedef __bf16 v16bf __attribute__((ext_vector_type(16)));
typedef float  v8f   __attribute__((ext_vector_type(8)));

// ---------------- 1x1 qkv conv (only the q/k/v slices actually used) -------
__global__ __launch_bounds__(256) void qkv_pointwise_kernel(
    const float* __restrict__ xq, const float* __restrict__ xkv,
    const float* __restrict__ w, float* __restrict__ tmp) {
  const int n   = blockIdx.x * 256 + threadIdx.x;   // 0..4095
  const int ocg = blockIdx.y;                       // 0..383
  const float* x  = (ocg < 128) ? xq : xkv;
  const float* wr = w + (size_t)ocg * 128;
  float acc = 0.f;
#pragma unroll 8
  for (int ic = 0; ic < 128; ++ic)
    acc = fmaf(wr[ic], x[(size_t)ic * N_TOK + n], acc);
  tmp[(size_t)ocg * N_TOK + n] = acc;
}

// ---------------- 3x3 depthwise, SAME pad, write bf16 [part][head][tok][d] --
__global__ __launch_bounds__(256) void dwconv_kernel(
    const float* __restrict__ tmp, const float* __restrict__ wdw,
    __bf16* __restrict__ qkvbf) {
  const int n  = blockIdx.x * 256 + threadIdx.x;    // token
  const int cg = blockIdx.y;                        // 0..383 global channel
  const int x0 = n & 63, y0 = n >> 6;
  const float* src = tmp + (size_t)cg * N_TOK;
  const float* wk  = wdw + (size_t)cg * 9;
  float acc = 0.f;
#pragma unroll
  for (int dy = -1; dy <= 1; ++dy)
#pragma unroll
    for (int dx = -1; dx <= 1; ++dx) {
      const int yy = y0 + dy, xx = x0 + dx;
      const float v = (yy >= 0 && yy < 64 && xx >= 0 && xx < 64)
                          ? src[yy * 64 + xx] : 0.f;
      acc = fmaf(wk[(dy + 1) * 3 + (dx + 1)], v, acc);
    }
  const int p = cg >> 7, c = cg & 127, hh = c >> 5, d = c & 31;
  qkvbf[((size_t)p * HEADS + hh) * (size_t)N_TOK * HD + (size_t)n * HD + d] =
      (__bf16)acc;
}

// ---------------- DPP 16-lane butterfly reductions (no LDS, no waits) -------
__device__ inline float dpp_bcast_xor1(float v) {
  return __int_as_float(__builtin_amdgcn_update_dpp(
      0, __float_as_int(v), 0xB1 /*quad_perm 1,0,3,2*/, 0xf, 0xf, true));
}
__device__ inline float dpp_bcast_xor2(float v) {
  return __int_as_float(__builtin_amdgcn_update_dpp(
      0, __float_as_int(v), 0x4E /*quad_perm 2,3,0,1*/, 0xf, 0xf, true));
}
__device__ inline float dpp_half_mirror(float v) {
  return __int_as_float(__builtin_amdgcn_update_dpp(
      0, __float_as_int(v), 0x141 /*row_half_mirror*/, 0xf, 0xf, true));
}
__device__ inline float dpp_mirror(float v) {
  return __int_as_float(__builtin_amdgcn_update_dpp(
      0, __float_as_int(v), 0x140 /*row_mirror*/, 0xf, 0xf, true));
}
__device__ inline float rowmax16(float v) {
  v = fmaxf(v, dpp_bcast_xor1(v));
  v = fmaxf(v, dpp_bcast_xor2(v));
  v = fmaxf(v, dpp_half_mirror(v));
  v = fmaxf(v, dpp_mirror(v));
  return v;
}
__device__ inline float rowsum16(float v) {
  v += dpp_bcast_xor1(v);
  v += dpp_bcast_xor2(v);
  v += dpp_half_mirror(v);
  v += dpp_mirror(v);
  return v;
}

// ---------------- flash attention, bf16 WMMA, f32 softmax -------------------
__global__ __launch_bounds__(256) void flash_attn_kernel(
    const __bf16* __restrict__ qbf, const __bf16* __restrict__ kbf,
    const __bf16* __restrict__ vbf, const float* __restrict__ temperature,
    float* __restrict__ out /* [C][N], c = h*HD+d */) {
  __shared__ __bf16 sK[2][KB][HD];        // [buf][key][d]
  __shared__ __bf16 sVt[2][HD][KB];       // [buf][d][key] (transposed)
  __shared__ __bf16 sP[WAVES][QW][KB];    // per-wave P staging (C->A relayout)

  const int h    = blockIdx.y;
  const int tid  = threadIdx.x;
  const int wave = tid >> 5;
  const int lane = tid & 31;
  const int half = lane >> 4;             // which 16-lane half
  const int l16  = lane & 15;
  const int qbase = blockIdx.x * (WAVES * QW) + wave * QW;
  const float temp = temperature[h];

  const __bf16* Qh = qbf + (size_t)h * N_TOK * HD;
  const __bf16* Kh = kbf + (size_t)h * N_TOK * HD;
  const __bf16* Vh = vbf + (size_t)h * N_TOK * HD;

  // stage one K/V tile into LDS buffer `buf` (register-buffered so the
  // global side vectorizes to b128 loads)
  auto stage = [&](int kb, int buf) {
    const int idx = tid * 8;
    const __bf16* ksrc = Kh + (size_t)kb * KB * HD + idx;
    const __bf16* vsrc = Vh + (size_t)kb * KB * HD + idx;
    __bf16 kv[8], vv[8];
#pragma unroll
    for (int j = 0; j < 8; ++j) { kv[j] = ksrc[j]; vv[j] = vsrc[j]; }
    __bf16* kd = &sK[buf][0][0] + idx;
#pragma unroll
    for (int j = 0; j < 8; ++j) kd[j] = kv[j];
#pragma unroll
    for (int j = 0; j < 8; ++j) {
      const int t = idx + j;
      sVt[buf][t & 31][t >> 5] = vv[j];
    }
  };

  // Q tile in A-matrix layout: lane -> M = l16; elems e -> K = half*16 + e.
  v16bf qA;
  {
    const __bf16* qrow = Qh + (size_t)(qbase + l16) * HD;
#pragma unroll
    for (int e = 0; e < 8; ++e) qA[e] = qrow[half * 16 + e];
#pragma unroll
    for (int e = 0; e < 8; ++e) qA[8 + e] = qrow[half * 16 + 8 + e];
  }

  v8f O0 = {}, O1 = {};          // output accum: d 0..15 and 16..31
  float mrow[8], lrow[8];        // per-row running max / sum (row = r+8*half)
#pragma unroll
  for (int r = 0; r < 8; ++r) { mrow[r] = -1e30f; lrow[r] = 0.f; }

  const int NIT = N_TOK / KB;
  stage(0, 0);                   // prologue

  for (int kb = 0; kb < NIT; ++kb) {
    const int cur = kb & 1;
    // Single barrier per iteration: publishes buffer `cur` (written last
    // iteration) AND guarantees buffer cur^1 is no longer being read, so we
    // can immediately start staging the next tile into it (overlapping the
    // global loads with this tile's WMMAs/softmax).
    __syncthreads();
    if (kb + 1 < NIT) stage(kb + 1, cur ^ 1);

    // K^T tiles in B layout: lane -> N = key = j*16 + l16,
    // elems e -> K(reduction d) = half*16 + e  (contiguous in sK rows).
    v16bf kB0, kB1, kB2, kB3;
#pragma unroll
    for (int e = 0; e < 16; ++e) {
      kB0[e] = sK[cur][l16][half * 16 + e];
      kB1[e] = sK[cur][16 + l16][half * 16 + e];
      kB2[e] = sK[cur][32 + l16][half * 16 + e];
      kB3[e] = sK[cur][48 + l16][half * 16 + e];
    }

    v8f S0 = {}, S1 = {}, S2 = {}, S3 = {};
    S0 = __builtin_amdgcn_wmma_f32_16x16x32_bf16(false, qA, false, kB0,
                                                 (short)0, S0, false, false);
    S1 = __builtin_amdgcn_wmma_f32_16x16x32_bf16(false, qA, false, kB1,
                                                 (short)0, S1, false, false);
    S2 = __builtin_amdgcn_wmma_f32_16x16x32_bf16(false, qA, false, kB2,
                                                 (short)0, S2, false, false);
    S3 = __builtin_amdgcn_wmma_f32_16x16x32_bf16(false, qA, false, kB3,
                                                 (short)0, S3, false, false);

    // online softmax per query row (row r+8*half; 16 N-values across lanes)
#pragma unroll
    for (int r = 0; r < 8; ++r) {
      const float s0 = S0[r] * temp, s1 = S1[r] * temp;
      const float s2 = S2[r] * temp, s3 = S3[r] * temp;
      const float mx   = rowmax16(fmaxf(fmaxf(s0, s1), fmaxf(s2, s3)));
      const float mnew = fmaxf(mrow[r], mx);
      const float resc = __expf(mrow[r] - mnew);
      const float p0 = __expf(s0 - mnew), p1 = __expf(s1 - mnew);
      const float p2 = __expf(s2 - mnew), p3 = __expf(s3 - mnew);
      lrow[r] = lrow[r] * resc + rowsum16((p0 + p1) + (p2 + p3));
      mrow[r] = mnew;
      O0[r] *= resc;
      O1[r] *= resc;
      // stage P (C layout) -> LDS [query][key] for C->A relayout.
      // Wave-local producer/consumer: no block barrier needed, the backend's
      // dscnt wait on the aliasing ds_load provides ordering.
      const int row = r + 8 * half;
      sP[wave][row][l16]      = (__bf16)p0;
      sP[wave][row][16 + l16] = (__bf16)p1;
      sP[wave][row][32 + l16] = (__bf16)p2;
      sP[wave][row][48 + l16] = (__bf16)p3;
    }

    // P tiles in A layout (key chunks 0..31 and 32..63)
    v16bf pA0, pA1;
    {
      const __bf16* prow = &sP[wave][l16][0];
#pragma unroll
      for (int e = 0; e < 8; ++e) {
        pA0[e]     = prow[half * 16 + e];
        pA0[8 + e] = prow[half * 16 + 8 + e];
        pA1[e]     = prow[32 + half * 16 + e];
        pA1[8 + e] = prow[40 + half * 16 + e];
      }
    }

    // V tiles in B layout: lane -> N = d = l16 (+16), elems e ->
    // K(reduction key) = chunk*32 + half*16 + e (contiguous in sVt rows).
    v16bf vB00, vB01, vB10, vB11;
#pragma unroll
    for (int e = 0; e < 16; ++e) {
      vB00[e] = sVt[cur][l16][half * 16 + e];
      vB01[e] = sVt[cur][16 + l16][half * 16 + e];
      vB10[e] = sVt[cur][l16][32 + half * 16 + e];
      vB11[e] = sVt[cur][16 + l16][32 + half * 16 + e];
    }

    O0 = __builtin_amdgcn_wmma_f32_16x16x32_bf16(false, pA0, false, vB00,
                                                 (short)0, O0, false, false);
    O0 = __builtin_amdgcn_wmma_f32_16x16x32_bf16(false, pA1, false, vB10,
                                                 (short)0, O0, false, false);
    O1 = __builtin_amdgcn_wmma_f32_16x16x32_bf16(false, pA0, false, vB01,
                                                 (short)0, O1, false, false);
    O1 = __builtin_amdgcn_wmma_f32_16x16x32_bf16(false, pA1, false, vB11,
                                                 (short)0, O1, false, false);
  }

  // normalize and scatter to [c][n] layout for the projection GEMM
#pragma unroll
  for (int r = 0; r < 8; ++r) {
    const int q = qbase + r + 8 * half;
    const float inv = 1.0f / lrow[r];
    out[(size_t)(h * HD + l16) * N_TOK + q]      = O0[r] * inv;
    out[(size_t)(h * HD + 16 + l16) * N_TOK + q] = O1[r] * inv;
  }
}

// ---------------- 1x1 projection -------------------------------------------
__global__ __launch_bounds__(256) void proj_kernel(
    const float* __restrict__ ain, const float* __restrict__ wp,
    float* __restrict__ out) {
  const int n  = blockIdx.x * 256 + threadIdx.x;
  const int oc = blockIdx.y;
  const float* wr = wp + (size_t)oc * 128;
  float acc = 0.f;
#pragma unroll 8
  for (int ic = 0; ic < 128; ++ic)
    acc = fmaf(wr[ic], ain[(size_t)ic * N_TOK + n], acc);
  out[(size_t)oc * N_TOK + n] = acc;
}

extern "C" void kernel_launch(void* const* d_in, const int* in_sizes, int n_in,
                              void* d_out, int out_size, void* d_ws,
                              size_t ws_size, hipStream_t stream) {
  const float* xq    = (const float*)d_in[0];
  const float* xkv   = (const float*)d_in[1];
  const float* wqkv  = (const float*)d_in[2];
  const float* wdw   = (const float*)d_in[3];
  const float* wproj = (const float*)d_in[4];
  const float* temp  = (const float*)d_in[5];
  float* out = (float*)d_out;

  // workspace layout (needs ~11.5 MB):
  //   tmp   : 3*128*4096 f32   (post-1x1, pre-depthwise)
  //   qkvbf : 3*128*4096 bf16  ([part][head][token][d])
  //   aout  : 128*4096   f32   (attention result, [c][n])
  char* ws = (char*)d_ws;
  float*  tmp   = (float*)ws;
  __bf16* qkvbf = (__bf16*)(ws + (size_t)3 * 128 * N_TOK * 4);
  float*  aout  = (float*)(ws + (size_t)3 * 128 * N_TOK * 4 +
                                (size_t)3 * 128 * N_TOK * 2);

  qkv_pointwise_kernel<<<dim3(16, 384), 256, 0, stream>>>(xq, xkv, wqkv, tmp);
  dwconv_kernel<<<dim3(16, 384), 256, 0, stream>>>(tmp, wdw, qkvbf);
  flash_attn_kernel<<<dim3(N_TOK / (WAVES * QW), HEADS), 256, 0, stream>>>(
      qkvbf,                                    // Q
      qkvbf + (size_t)128 * N_TOK,              // K
      qkvbf + (size_t)2 * 128 * N_TOK,          // V
      temp, aout);
  proj_kernel<<<dim3(16, 128), 256, 0, stream>>>(aout, wproj, out);
}
